// ProteinGNN_48275432407807
// MI455X (gfx1250) — matrix operand order repaired and checked by reading, not compiled, not run on hardware
//
#include <hip/hip_runtime.h>
#include <cstdint>
#include <cstddef>

typedef __attribute__((ext_vector_type(16))) __bf16 v16bf;
typedef __attribute__((ext_vector_type(8)))  __bf16 v8bf;
typedef __attribute__((ext_vector_type(4)))  __bf16 bf16x4;
typedef __attribute__((ext_vector_type(8)))  float  v8f;

union ABfrag { v16bf v; v8bf h[2]; };

#define N_AA      300000
#define N_PROT    30000
#define DFEAT     128
#define HID       128
#define NOUT      64
#define E_AP      300000
#define E_PP      960000
#define BATCH     16384
#define NEG_SLOPE 0.2f
#define NTHR      256     // 8 waves per block

// ---------------------------------------------------------------------------
// Workspace float offsets
// ---------------------------------------------------------------------------
#define OFF_XL_AP   ((size_t)0)                                  // 300000*128
#define OFF_XR_AP   (OFF_XL_AP + (size_t)N_AA * HID)             // 30000*128
#define OFF_XL_PP   (OFF_XR_AP + (size_t)N_PROT * HID)
#define OFF_XR_PP   (OFF_XL_PP + (size_t)N_PROT * HID)
#define OFF_ACC     (OFF_XR_PP + (size_t)N_PROT * HID)           // 30000*128
#define OFF_LMAX_AP (OFF_ACC + (size_t)N_PROT * HID)
#define OFF_DEN_AP  (OFF_LMAX_AP + N_PROT)
#define OFF_LMAX_PP (OFF_DEN_AP + N_PROT)
#define OFF_DEN_PP  (OFF_LMAX_PP + N_PROT)
#define OFF_E_AP    (OFF_DEN_PP + N_PROT)                        // 300000
#define OFF_E_PP    (OFF_E_AP + E_AP)                            // 960000

// XOR swizzle of the 8-element chunk index with the row id: keeps every
// 16-byte fragment load (ds_load_b128) conflict-free across the 16 lanes
// of a half-wave. k in [0,128), r = row & 15.
__device__ __forceinline__ int swz(int k, int r) {
  return (k & 7) | ((((k >> 3) ^ r) & 15) << 3);
}

// ---------------------------------------------------------------------------
// init: zero the segment accumulator / denominators, set lmax = -inf
// ---------------------------------------------------------------------------
__global__ void init_kernel(float* __restrict__ acc,
                            float* __restrict__ lmax_ap, float* __restrict__ den_ap,
                            float* __restrict__ lmax_pp, float* __restrict__ den_pp) {
  int i = blockIdx.x * blockDim.x + threadIdx.x;
  if (i < N_PROT * HID) acc[i] = 0.0f;
  if (i < N_PROT) {
    lmax_ap[i] = -__builtin_inff();
    lmax_pp[i] = -__builtin_inff();
    den_ap[i]  = 0.0f;
    den_pp[i]  = 0.0f;
  }
}

// ---------------------------------------------------------------------------
// Shared WMMA inner loop: one wave computes a 16 x NCOL output stripe from a
// 16x128 bf16 A tile (xrow, swizzled) and a transposed+swizzled 128xNCOL
// weight tile sW[n*128 + swz(k,n&15)].
// A-frag (16x32 bf16): 16 contiguous K elems per lane (k = 16h + s), same order as B.
// B-frag (32x16 bf16): 16 contiguous K elems per lane in transposed W.
// ---------------------------------------------------------------------------
template<int NCOL>
__device__ __forceinline__ void wmma_stripe(
    const __bf16* __restrict__ xrow, const __bf16* __restrict__ sW,
    int lane, v8f (&acc)[NCOL / 16])
{
  const int halfA = (lane >> 4) << 3;   // +8 K-offset, unused (A uses halfB)
  const int halfB = (lane >> 4) << 4;   // +16 K-offset for B (lanes 16-31)
  const int m     = lane & 15;
  const int nloc  = lane & 15;

  #pragma unroll
  for (int kc = 0; kc < 4; ++kc) {
    const int kb = kc * 32;
    ABfrag a;
    a.h[0] = *(const v8bf*)(xrow + swz(kb + halfB, m));
    a.h[1] = *(const v8bf*)(xrow + swz(kb + halfB + 8, m));
    #pragma unroll
    for (int nt = 0; nt < NCOL / 16; ++nt) {
      const __bf16* wcol = sW + (size_t)(nt * 16 + nloc) * 128;
      ABfrag b;
      b.h[0] = *(const v8bf*)(wcol + swz(kb + halfB, nloc));
      b.h[1] = *(const v8bf*)(wcol + swz(kb + halfB + 8, nloc));
      acc[nt] = __builtin_amdgcn_wmma_f32_16x16x32_bf16(
          false, a.v, false, b.v, (short)0, acc[nt], false, false);
    }
  }
}

// ---------------------------------------------------------------------------
// WMMA transform: Y = X @ W + b   (X: nrows x 128 f32, W: 128 x NCOL)
// 8 waves / block, one 16-row tile per wave.
// ---------------------------------------------------------------------------
template<int NCOL>
__global__ __launch_bounds__(NTHR) void xform_kernel(
    const float* __restrict__ X, const float* __restrict__ W,
    const float* __restrict__ bias, float* __restrict__ Y, int nrows)
{
  __shared__ __bf16 sW[(size_t)NCOL * 128];   // transposed + swizzled
  __shared__ float  sBias[NCOL];
  __shared__ __bf16 sX[8][16 * 128];          // swizzled, one tile per wave

  const int tid  = threadIdx.x;
  const int wave = tid >> 5;
  const int lane = tid & 31;

  // stage weights: W[k][n] -> sW[n*128 + swz(k, n&15)]  (bf16)
  for (int i = tid; i < 128 * NCOL; i += NTHR) {
    int k = i / NCOL, n = i % NCOL;
    sW[(size_t)n * 128 + swz(k, n & 15)] = (__bf16)W[i];
  }
  if (tid < NCOL) sBias[tid] = bias[tid];

  const int  tile    = blockIdx.x * 8 + wave;
  const bool active  = (tile * 16) < nrows;
  const int  rowBase = active ? tile * 16 : 0;

  // stage 16x128 X tile: coalesced float4 loads, bf16x4 swizzled LDS stores
  const float4* Xr4 = (const float4*)(X + (size_t)rowBase * 128);
  #pragma unroll
  for (int j = 0; j < 16; ++j) {
    int idx = j * 32 + lane;        // float4 index in tile (512 total)
    int m   = idx >> 5;             // 32 float4 per row
    int k   = (idx & 31) << 2;
    float4 f = Xr4[idx];
    bf16x4 p;
    p.x = (__bf16)f.x; p.y = (__bf16)f.y; p.z = (__bf16)f.z; p.w = (__bf16)f.w;
    *(bf16x4*)(&sX[wave][m * 128 + swz(k, m)]) = p;
  }
  __syncthreads();

  v8f acc[NCOL / 16] = {};
  wmma_stripe<NCOL>(&sX[wave][(lane & 15) * 128], sW, lane, acc);

  if (active) {
    const int rOff = (lane >> 4) << 3;
    const int nloc = lane & 15;
    #pragma unroll
    for (int nt = 0; nt < NCOL / 16; ++nt)
      #pragma unroll
      for (int r = 0; r < 8; ++r) {
        int row = rowBase + r + rOff;
        int col = nt * 16 + nloc;
        Y[(size_t)row * NCOL + col] = acc[nt][r] + sBias[col];
      }
  }
}

// ---------------------------------------------------------------------------
// Edge phase 1: logits[e] = att . leaky_relu(xl[src]+xr[dst]); atomic seg-max
// ---------------------------------------------------------------------------
__device__ __forceinline__ void atomicMaxF32(float* addr, float val) {
  if (val >= 0.0f)
    atomicMax((int*)addr, __float_as_int(val));
  else
    atomicMin((unsigned int*)addr, (unsigned int)__float_as_int(val));
}

__global__ __launch_bounds__(256) void edge_logits_kernel(
    const float* __restrict__ xl, const float* __restrict__ xr,
    const int* __restrict__ src, const int* __restrict__ dst,
    const float* __restrict__ att, float* __restrict__ logits,
    float* __restrict__ lmax, int E)
{
  int e    = blockIdx.x * 8 + (threadIdx.x >> 5);
  int lane = threadIdx.x & 31;
  if (e >= E) return;
  int s = src[e], d = dst[e];
  float4 a = ((const float4*)(xl + (size_t)s * HID))[lane];
  float4 b = ((const float4*)(xr + (size_t)d * HID))[lane];
  float4 w = ((const float4*)att)[lane];
  float v0 = a.x + b.x; v0 = v0 > 0.0f ? v0 : NEG_SLOPE * v0;
  float v1 = a.y + b.y; v1 = v1 > 0.0f ? v1 : NEG_SLOPE * v1;
  float v2 = a.z + b.z; v2 = v2 > 0.0f ? v2 : NEG_SLOPE * v2;
  float v3 = a.w + b.w; v3 = v3 > 0.0f ? v3 : NEG_SLOPE * v3;
  float sum = v0 * w.x + v1 * w.y + v2 * w.z + v3 * w.w;
  #pragma unroll
  for (int off = 16; off > 0; off >>= 1) sum += __shfl_down(sum, off, 32);
  if (lane == 0) {
    logits[e] = sum;
    atomicMaxF32(&lmax[d], sum);
  }
}

// ---------------------------------------------------------------------------
// Edge phase 2: ex = exp(logit - lmax[dst]) (in place); denom[dst] += ex
// ---------------------------------------------------------------------------
__global__ __launch_bounds__(256) void edge_exp_kernel(
    float* __restrict__ elog, const int* __restrict__ dst,
    const float* __restrict__ lmax, float* __restrict__ denom, int E)
{
  int e = blockIdx.x * blockDim.x + threadIdx.x;
  if (e >= E) return;
  int d = dst[e];
  float ex = __expf(elog[e] - lmax[d]);
  elog[e] = ex;
  atomicAdd(&denom[d], ex);
}

// ---------------------------------------------------------------------------
// Edge phase 3: acc[dst,:] += (ex/denom[dst]) * xl[src,:]
// ---------------------------------------------------------------------------
__global__ __launch_bounds__(256) void edge_scatter_kernel(
    const float* __restrict__ xl, const float* __restrict__ ex,
    const float* __restrict__ denom, const int* __restrict__ src,
    const int* __restrict__ dst, float* __restrict__ acc, int E)
{
  int e    = blockIdx.x * 8 + (threadIdx.x >> 5);
  int lane = threadIdx.x & 31;
  if (e >= E) return;
  int s = src[e], d = dst[e];
  float alpha = ex[e] / denom[d];
  float4 v = ((const float4*)(xl + (size_t)s * HID))[lane];
  float* p = acc + (size_t)d * HID + lane * 4;
  atomicAdd(p + 0, alpha * v.x);
  atomicAdd(p + 1, alpha * v.y);
  atomicAdd(p + 2, alpha * v.z);
  atomicAdd(p + 3, alpha * v.w);
}

// ---------------------------------------------------------------------------
// Head: out = relu(acc + bias_ap + bias_pp)[:BATCH] @ W_lin + b_lin   (WMMA)
// ---------------------------------------------------------------------------
__global__ __launch_bounds__(NTHR) void head_kernel(
    const float* __restrict__ ACC, const float* __restrict__ bias_ap,
    const float* __restrict__ bias_pp, const float* __restrict__ W,
    const float* __restrict__ blin, float* __restrict__ Y, int nrows)
{
  __shared__ __bf16 sW[(size_t)NOUT * 128];   // transposed + swizzled
  __shared__ float  sBias[NOUT];
  __shared__ __bf16 sX[8][16 * 128];

  const int tid  = threadIdx.x;
  const int wave = tid >> 5;
  const int lane = tid & 31;

  for (int i = tid; i < 128 * NOUT; i += NTHR) {
    int k = i / NOUT, n = i % NOUT;
    sW[(size_t)n * 128 + swz(k, n & 15)] = (__bf16)W[i];
  }
  if (tid < NOUT) sBias[tid] = blin[tid];

  const int  tile    = blockIdx.x * 8 + wave;
  const bool active  = (tile * 16) < nrows;
  const int  rowBase = active ? tile * 16 : 0;

  const float4* Ar4 = (const float4*)(ACC + (size_t)rowBase * HID);
  const float4* Ba4 = (const float4*)bias_ap;
  const float4* Bp4 = (const float4*)bias_pp;
  #pragma unroll
  for (int j = 0; j < 16; ++j) {
    int idx = j * 32 + lane;
    int m   = idx >> 5;
    int k4  = idx & 31;
    int k   = k4 << 2;
    float4 f = Ar4[idx];
    float4 ba = Ba4[k4];
    float4 bp = Bp4[k4];
    float x0 = f.x + ba.x + bp.x; x0 = x0 > 0.0f ? x0 : 0.0f;
    float x1 = f.y + ba.y + bp.y; x1 = x1 > 0.0f ? x1 : 0.0f;
    float x2 = f.z + ba.z + bp.z; x2 = x2 > 0.0f ? x2 : 0.0f;
    float x3 = f.w + ba.w + bp.w; x3 = x3 > 0.0f ? x3 : 0.0f;
    bf16x4 p;
    p.x = (__bf16)x0; p.y = (__bf16)x1; p.z = (__bf16)x2; p.w = (__bf16)x3;
    *(bf16x4*)(&sX[wave][m * 128 + swz(k, m)]) = p;
  }
  __syncthreads();

  v8f acc[NOUT / 16] = {};
  wmma_stripe<NOUT>(&sX[wave][(lane & 15) * 128], sW, lane, acc);

  if (active) {
    const int rOff = (lane >> 4) << 3;
    const int nloc = lane & 15;
    #pragma unroll
    for (int nt = 0; nt < NOUT / 16; ++nt)
      #pragma unroll
      for (int r = 0; r < 8; ++r) {
        int row = rowBase + r + rOff;
        int col = nt * 16 + nloc;
        Y[(size_t)row * NOUT + col] = acc[nt][r] + sBias[col];
      }
  }
}

// ---------------------------------------------------------------------------
extern "C" void kernel_launch(void* const* d_in, const int* in_sizes, int n_in,
                              void* d_out, int out_size, void* d_ws, size_t ws_size,
                              hipStream_t stream) {
  const float* x_aa    = (const float*)d_in[0];
  const float* x_prot  = (const float*)d_in[1];
  const int*   ei_ap   = (const int*)d_in[2];
  const int*   ei_pp   = (const int*)d_in[3];
  // d_in[4] = batch_size (compile-time BATCH)
  const float* Wl_ap   = (const float*)d_in[5];
  const float* bl_ap   = (const float*)d_in[6];
  const float* Wr_ap   = (const float*)d_in[7];
  const float* br_ap   = (const float*)d_in[8];
  const float* att_ap  = (const float*)d_in[9];
  const float* bias_ap = (const float*)d_in[10];
  const float* Wl_pp   = (const float*)d_in[11];
  const float* bl_pp   = (const float*)d_in[12];
  const float* Wr_pp   = (const float*)d_in[13];
  const float* br_pp   = (const float*)d_in[14];
  const float* att_pp  = (const float*)d_in[15];
  const float* bias_pp = (const float*)d_in[16];
  const float* W_lin   = (const float*)d_in[17];
  const float* b_lin   = (const float*)d_in[18];

  float* ws      = (float*)d_ws;
  float* xl_ap   = ws + OFF_XL_AP;
  float* xr_ap   = ws + OFF_XR_AP;
  float* xl_pp   = ws + OFF_XL_PP;
  float* xr_pp   = ws + OFF_XR_PP;
  float* acc     = ws + OFF_ACC;
  float* lmax_ap = ws + OFF_LMAX_AP;
  float* den_ap  = ws + OFF_DEN_AP;
  float* lmax_pp = ws + OFF_LMAX_PP;
  float* den_pp  = ws + OFF_DEN_PP;
  float* e_ap    = ws + OFF_E_AP;
  float* e_pp    = ws + OFF_E_PP;

  const int* src_ap = ei_ap;
  const int* dst_ap = ei_ap + E_AP;
  const int* src_pp = ei_pp;
  const int* dst_pp = ei_pp + E_PP;

  // 0) init accumulator / softmax state
  init_kernel<<<(N_PROT * HID + 255) / 256, 256, 0, stream>>>(
      acc, lmax_ap, den_ap, lmax_pp, den_pp);

  // 1) node transforms (WMMA bf16, 8 tiles per block)
  {
    int blocks;
    blocks = (N_AA / 16 + 7) / 8;
    xform_kernel<HID><<<blocks, NTHR, 0, stream>>>(x_aa,   Wl_ap, bl_ap, xl_ap, N_AA);
    blocks = (N_PROT / 16 + 7) / 8;
    xform_kernel<HID><<<blocks, NTHR, 0, stream>>>(x_prot, Wr_ap, br_ap, xr_ap, N_PROT);
    xform_kernel<HID><<<blocks, NTHR, 0, stream>>>(x_prot, Wl_pp, bl_pp, xl_pp, N_PROT);
    xform_kernel<HID><<<blocks, NTHR, 0, stream>>>(x_prot, Wr_pp, br_pp, xr_pp, N_PROT);
  }

  // 2) edge logits + segment max
  edge_logits_kernel<<<(E_AP + 7) / 8, 256, 0, stream>>>(
      xl_ap, xr_ap, src_ap, dst_ap, att_ap, e_ap, lmax_ap, E_AP);
  edge_logits_kernel<<<(E_PP + 7) / 8, 256, 0, stream>>>(
      xl_pp, xr_pp, src_pp, dst_pp, att_pp, e_pp, lmax_pp, E_PP);

  // 3) exp + segment denominator
  edge_exp_kernel<<<(E_AP + 255) / 256, 256, 0, stream>>>(
      e_ap, dst_ap, lmax_ap, den_ap, E_AP);
  edge_exp_kernel<<<(E_PP + 255) / 256, 256, 0, stream>>>(
      e_pp, dst_pp, lmax_pp, den_pp, E_PP);

  // 4) weighted scatter of both convs into one accumulator (x = out_ap+out_pp)
  edge_scatter_kernel<<<(E_AP + 7) / 8, 256, 0, stream>>>(
      xl_ap, e_ap, den_ap, src_ap, dst_ap, acc, E_AP);
  edge_scatter_kernel<<<(E_PP + 7) / 8, 256, 0, stream>>>(
      xl_pp, e_pp, den_pp, src_pp, dst_pp, acc, E_PP);

  // 5) relu + linear head (WMMA) -> d_out [BATCH x NOUT]
  head_kernel<<<(BATCH / 16 + 7) / 8, NTHR, 0, stream>>>(
      acc, bias_ap, bias_pp, W_lin, b_lin, (float*)d_out, BATCH);
}
